// SelfGoverningVacancy_81312320848235
// MI455X (gfx1250) — compile-verified, matrix operand order checked by reading, not run on hardware
//
#include <hip/hip_runtime.h>
#include <hip/hip_bf16.h>

typedef __attribute__((ext_vector_type(16))) __bf16 v16bf;
typedef __attribute__((ext_vector_type(8)))  __bf16 v8bf;
typedef __attribute__((ext_vector_type(8)))  float  v8f;

#define N_TOT   65536      // 64*32*32
#define DLAT    256
#define KCB     1024
#define HW      1024       // 32*32

// ---------------------------------------------------------------------------
// K1: (B,d,h,w) fp32 -> (N,d) bf16 row-major, LDS-tiled transpose.
// Block handles 32 channels x 32 x-positions at fixed (b, y).
// ---------------------------------------------------------------------------
__global__ __launch_bounds__(256) void k_transpose_bf16(
    const float* __restrict__ z, __bf16* __restrict__ zb)
{
    __shared__ float tile[32][33];
    const int tb = blockIdx.x;
    const int ct = tb & 7;          // channel tile (8 tiles of 32)
    const int y  = (tb >> 3) & 31;
    const int b  = tb >> 8;
    const int t  = threadIdx.x;
#pragma unroll
    for (int j = 0; j < 4; ++j) {
        int e = t + j * 256;
        int cl = e >> 5, x = e & 31;
        tile[cl][x] = z[(((size_t)b * 256 + ct * 32 + cl) * 32 + y) * 32 + x];
    }
    __syncthreads();
#pragma unroll
    for (int j = 0; j < 4; ++j) {
        int e = t + j * 256;
        int xl = e >> 5, cl = e & 31;
        zb[(((size_t)b * 32 + y) * 32 + xl) * 256 + ct * 32 + cl] =
            (__bf16)tile[cl][xl];
    }
}

// ---------------------------------------------------------------------------
// K2: codebook fp32 -> bf16, plus per-code squared norm.
// ---------------------------------------------------------------------------
__global__ __launch_bounds__(256) void k_cb_prep(
    const float* __restrict__ cb, __bf16* __restrict__ cbb,
    float* __restrict__ cbn)
{
    __shared__ float red[256];
    const int k = blockIdx.x, c = threadIdx.x;
    float v = cb[k * 256 + c];
    cbb[k * 256 + c] = (__bf16)v;
    red[c] = v * v;
    __syncthreads();
    for (int s = 128; s > 0; s >>= 1) {
        if (c < s) red[c] += red[c + s];
        __syncthreads();
    }
    if (c == 0) cbn[k] = red[0];
}

// ---------------------------------------------------------------------------
// K3: nearest-code search. One wave = 16 rows; A fragments held in VGPRs,
// 64 code tiles x 8 chained v_wmma_f32_16x16x32_bf16, fused score + argmin.
// ---------------------------------------------------------------------------
__global__ __launch_bounds__(256) void k_vq_argmin(
    const __bf16* __restrict__ zb,    // (N,256) bf16
    const __bf16* __restrict__ cbb,   // (1024,256) bf16
    const float*  __restrict__ cbn,   // (1024) ||cb||^2
    int*   __restrict__ idx_out,      // (N) int
    float* __restrict__ idx_out_f)    // (N) float (d_out indices slot)
{
    const int lane = threadIdx.x & 31;
    const int wave = threadIdx.x >> 5;
    const int rowBase = blockIdx.x * 128 + wave * 16;
    const int r  = lane & 15;                 // row (A) / col (B) within tile
    const int kb = (lane < 16) ? 0 : 8;       // K sub-base per ISA layout

    // A fragments: row rowBase+r, two 16B chunks per 32-wide k-step.
    v16bf aF[8];
    const __bf16* zrow = zb + (size_t)(rowBase + r) * 256;
#pragma unroll
    for (int ks = 0; ks < 8; ++ks) {
        v8bf lo = *(const v8bf*)(zrow + ks * 32 + kb);
        v8bf hi = *(const v8bf*)(zrow + ks * 32 + kb + 16);
        aF[ks] = __builtin_shufflevector(lo, hi,
            0,1,2,3,4,5,6,7,8,9,10,11,12,13,14,15);
    }

    float bestVal[8];
    int   bestIdx[8];
#pragma unroll
    for (int i = 0; i < 8; ++i) { bestVal[i] = 3.0e38f; bestIdx[i] = 0; }

    for (int t = 0; t < 64; ++t) {
        const int code = t * 16 + r;
        const __bf16* crow = cbb + (size_t)code * 256;
        const float cn = cbn[code];
        v8f acc = {};
#pragma unroll
        for (int ks = 0; ks < 8; ++ks) {
            v8bf lo = *(const v8bf*)(crow + ks * 32 + kb);
            v8bf hi = *(const v8bf*)(crow + ks * 32 + kb + 16);
            v16bf bF = __builtin_shufflevector(lo, hi,
                0,1,2,3,4,5,6,7,8,9,10,11,12,13,14,15);
            acc = __builtin_amdgcn_wmma_f32_16x16x32_bf16(
                false, aF[ks], false, bF, (short)0, acc, false, false);
        }
#pragma unroll
        for (int i = 0; i < 8; ++i) {
            float s = cn - 2.0f * acc[i];     // ||cb||^2 - 2 z.cb
            if (s < bestVal[i]) { bestVal[i] = s; bestIdx[i] = code; }
        }
    }

    // Reduce across the 16 lanes that share each row (masks stay in-half).
#pragma unroll
    for (int i = 0; i < 8; ++i) {
#pragma unroll
        for (int m = 1; m < 16; m <<= 1) {
            float ov = __shfl_xor(bestVal[i], m, 32);
            int   oi = __shfl_xor(bestIdx[i], m, 32);
            if (ov < bestVal[i] || (ov == bestVal[i] && oi < bestIdx[i])) {
                bestVal[i] = ov; bestIdx[i] = oi;
            }
        }
    }
    if (r == 0) {
        const int rowOff = (lane >= 16) ? 8 : 0;   // lanes 16-31 hold M=8..15
#pragma unroll
        for (int i = 0; i < 8; ++i) {
            int row = rowBase + rowOff + i;
            idx_out[row]   = bestIdx[i];
            idx_out_f[row] = (float)bestIdx[i];
        }
    }
}

// ---------------------------------------------------------------------------
// K4: gather codebook rows, write e_k_ste (== e_k) and delta = z - e_k,
// in (B,d,h,w) layout, fully coalesced on z/outputs; gathers hit L2.
// ---------------------------------------------------------------------------
__global__ __launch_bounds__(256) void k_gather_out(
    const float* __restrict__ z, const float* __restrict__ cb,
    const int* __restrict__ idx,
    float* __restrict__ ek, float* __restrict__ delta)
{
    const int base = blockIdx.x * 1024 + threadIdx.x;
#pragma unroll
    for (int j = 0; j < 4; ++j) {
        int o = base + j * 256;
        int x = o & 31, y = (o >> 5) & 31, c = (o >> 10) & 255, b = o >> 18;
        int n = b * HW + y * 32 + x;
        float e  = cb[idx[n] * 256 + c];
        float zv = z[o];
        ek[o]    = e;
        delta[o] = zv - e;
    }
}

// ---------------------------------------------------------------------------
// K5: scatter counts & sums (one block per flattened position n).
// ---------------------------------------------------------------------------
__global__ __launch_bounds__(256) void k_scatter(
    const float* __restrict__ z, const int* __restrict__ idx,
    float* __restrict__ wcount, float* __restrict__ wsum)
{
    const int n = blockIdx.x, c = threadIdx.x;
    const int b = n >> 10, rem = n & 1023;
    const int k = idx[n];
    float zv = z[(size_t)b * 262144 + (size_t)c * 1024 + rem];
    atomicAdd(&wsum[k * 256 + c], zv);
    if (c == 0) atomicAdd(&wcount[k], 1.0f);
}

// ---------------------------------------------------------------------------
// K6a: ema_count update + total n reduction.
// ---------------------------------------------------------------------------
__global__ __launch_bounds__(1024) void k_ema_count(
    const float* __restrict__ ema_count, const float* __restrict__ wcount,
    float* __restrict__ out_count, float* __restrict__ nsum)
{
    __shared__ float red[1024];
    const int i = threadIdx.x;
    float ec = 0.99f * ema_count[i] + 0.01f * wcount[i];
    out_count[i] = ec;
    red[i] = ec;
    __syncthreads();
    for (int s = 512; s > 0; s >>= 1) {
        if (i < s) red[i] += red[i + s];
        __syncthreads();
    }
    if (i == 0) nsum[0] = red[0];
}

// ---------------------------------------------------------------------------
// K6b: ema_sum update + new codebook.
// ---------------------------------------------------------------------------
__global__ __launch_bounds__(256) void k_ema_finish(
    const float* __restrict__ ema_sum, const float* __restrict__ wsum,
    const float* __restrict__ out_count, const float* __restrict__ nsum,
    float* __restrict__ out_sum, float* __restrict__ out_cb)
{
    const int k = blockIdx.x, c = threadIdx.x;
    float es = 0.99f * ema_sum[k * 256 + c] + 0.01f * wsum[k * 256 + c];
    out_sum[k * 256 + c] = es;
    float n  = nsum[0];
    float ec = out_count[k];
    float cs = (ec + 1e-5f) / (n + 1024.0f * 1e-5f) * n;
    out_cb[k * 256 + c] = es / cs;
}

// ---------------------------------------------------------------------------
extern "C" void kernel_launch(void* const* d_in, const int* in_sizes, int n_in,
                              void* d_out, int out_size, void* d_ws, size_t ws_size,
                              hipStream_t stream)
{
    const float* z_e       = (const float*)d_in[0];   // (64,256,32,32)
    const float* codebook  = (const float*)d_in[1];   // (1024,256)
    const float* ema_count = (const float*)d_in[2];   // (1024,)
    const float* ema_sum   = (const float*)d_in[3];   // (1024,256)

    float* out       = (float*)d_out;
    float* out_ek    = out;                   // 16777216
    float* out_idx   = out + 16777216;        // 65536
    float* out_delta = out + 16842752;        // 16777216
    float* out_count = out + 33619968;        // 1024
    float* out_sum   = out + 33620992;        // 262144
    float* out_cb    = out + 33883136;        // 262144

    char* ws = (char*)d_ws;
    __bf16* zb     = (__bf16*)(ws);                     // 33,554,432 B
    __bf16* cbb    = (__bf16*)(ws + 33554432);          //    524,288 B
    float*  cbn    = (float*)(ws + 34078720);           //      4,096 B
    int*    idxb   = (int*)  (ws + 34082816);           //    262,144 B
    float*  wcount = (float*)(ws + 34344960);           //      4,096 B
    float*  wsum   = (float*)(ws + 34349056);           //  1,048,576 B
    float*  nsum   = (float*)(ws + 35397632);           //          4 B

    hipMemsetAsync(wcount, 0, 4096, stream);
    hipMemsetAsync(wsum,   0, 1048576, stream);

    k_transpose_bf16<<<16384, 256, 0, stream>>>(z_e, zb);
    k_cb_prep      <<<1024,  256, 0, stream>>>(codebook, cbb, cbn);
    k_vq_argmin    <<<512,   256, 0, stream>>>(zb, cbb, cbn, idxb, out_idx);
    k_gather_out   <<<16384, 256, 0, stream>>>(z_e, codebook, idxb, out_ek, out_delta);
    k_scatter      <<<65536, 256, 0, stream>>>(z_e, idxb, wcount, wsum);
    k_ema_count    <<<1,    1024, 0, stream>>>(ema_count, wcount, out_count, nsum);
    k_ema_finish   <<<1024,  256, 0, stream>>>(ema_sum, wsum, out_count, nsum, out_sum, out_cb);
}
